// Actor_59614146068989
// MI455X (gfx1250) — matrix operand order repaired and checked by reading, not compiled, not run on hardware
//
#include <hip/hip_runtime.h>
#include <math.h>

typedef __attribute__((ext_vector_type(16))) __bf16 v16bf;
typedef __attribute__((ext_vector_type(8)))  __bf16 v8bf;
typedef __attribute__((ext_vector_type(8)))  float  v8f;

#define NB   128          // batch (graphs)
#define NPG  500          // nodes per graph
#define NN   (NB*NPG)     // 64000 nodes
#define HD   128
#define NEGV (-1e9f)

// ---------------------------------------------------------------------------
// WMMA helper: D = A(16x32 bf16) * B(32x16 bf16) + C(16x16 f32)
// ---------------------------------------------------------------------------
__device__ __forceinline__ v8f wmma_bf16(v16bf a, v16bf b, v8f c) {
    return __builtin_amdgcn_wmma_f32_16x16x32_bf16(
        /*neg_a=*/false, a, /*neg_b=*/false, b,
        /*c_mod=*/(short)0, c, /*reuse_a=*/false, /*reuse_b=*/false);
}

// ---------------------------------------------------------------------------
// Generic M=64000, N=128 GEMM: C = act(A[f32, MxK] @ W + bias), W given as
// Bt[bf16, 128xK] (transposed). One wave per 16-row M tile; loops all 8
// N-tiles reusing the A fragment. K-loop fully unrolled; all fragment loads
// for a K-block are issued before the 8-WMMA chain so the backend can clause
// the b128 loads and overlap them with matrix ops.
// ACT: 0 = none, 1 = tanh ; POOL: accumulate into npool
// ---------------------------------------------------------------------------
template <int K, int ACT, int POOL>
__global__ void gemm_wmma_128(const float* __restrict__ A,
                              const __bf16* __restrict__ Bt,
                              const float* __restrict__ bias,
                              float* __restrict__ C,
                              float* __restrict__ npool)
{
    const int wave  = threadIdx.x >> 5;
    const int lane  = threadIdx.x & 31;
    const int half  = lane >> 4;      // K sub-block selector
    const int r     = lane & 15;      // row (A) / col (B,C) within tile
    const int tileM = blockIdx.x * 4 + wave;

    const long rowA = (long)tileM * 16 + r;

    v8f acc[8];
    v8f zero = {};
#pragma unroll
    for (int t = 0; t < 8; ++t) acc[t] = zero;

#pragma unroll
    for (int kb = 0; kb < K; kb += 32) {
        // ---- issue all loads for this K block first -------------------
        const float* ar = A + rowA * K + kb;
        float4 a0 = *(const float4*)(ar + half * 16);
        float4 a1 = *(const float4*)(ar + half * 16 + 4);
        float4 a2 = *(const float4*)(ar + half * 16 + 8);
        float4 a3 = *(const float4*)(ar + half * 16 + 12);

        v16bf bfr[8];
#pragma unroll
        for (int tn = 0; tn < 8; ++tn) {
            const __bf16* br = Bt + (long)(tn * 16 + r) * K + kb + half * 16;
            bfr[tn] = *(const v16bf*)br;
        }

        // ---- convert A fragment (contiguous k = 16*half + s, same order as B) ---------------
        v16bf af;
        af[0]=(__bf16)a0.x; af[1]=(__bf16)a0.y; af[2]=(__bf16)a0.z; af[3]=(__bf16)a0.w;
        af[4]=(__bf16)a1.x; af[5]=(__bf16)a1.y; af[6]=(__bf16)a1.z; af[7]=(__bf16)a1.w;
        af[8]=(__bf16)a2.x; af[9]=(__bf16)a2.y; af[10]=(__bf16)a2.z; af[11]=(__bf16)a2.w;
        af[12]=(__bf16)a3.x; af[13]=(__bf16)a3.y; af[14]=(__bf16)a3.z; af[15]=(__bf16)a3.w;

        // ---- 8 back-to-back WMMAs ------------------------------------
#pragma unroll
        for (int tn = 0; tn < 8; ++tn)
            acc[tn] = wmma_bf16(af, bfr[tn], acc[tn]);
    }

#pragma unroll
    for (int tn = 0; tn < 8; ++tn) {
        const int col = tn * 16 + r;
        const float bv = bias[col];
#pragma unroll
        for (int j = 0; j < 8; ++j) {
            const long row = (long)tileM * 16 + half * 8 + j;
            float v = acc[tn][j] + bv;
            if (ACT == 1) v = tanhf(v);
            C[row * HD + col] = v;
            if (POOL) npool[row * HD + col] += v;   // unique element per lane
        }
    }
}

// ---------------------------------------------------------------------------
// Scores: S[g] = a_pad[g] (512x128 bf16) @ a_pad[g]^T ; mask + store 500x500.
// K = 128 fully unrolled; all fragments loaded before the WMMA chain.
// ---------------------------------------------------------------------------
__global__ void scores_wmma(const __bf16* __restrict__ apad,
                            const unsigned char* __restrict__ mask,
                            float* __restrict__ S)
{
    const int wid  = blockIdx.x * 4 + (threadIdx.x >> 5);
    const int lane = threadIdx.x & 31;
    const int half = lane >> 4;
    const int r    = lane & 15;
    const int g  = wid >> 10;
    const int t  = wid & 1023;
    const int tm = t >> 5;
    const int tn = t & 31;

    const __bf16* base = apad + (long)g * 512 * HD;
    const __bf16* arow = base + (long)(tm * 16 + r) * HD;
    const __bf16* brow = base + (long)(tn * 16 + r) * HD;

    v16bf afr[4], bfr[4];
#pragma unroll
    for (int q = 0; q < 4; ++q) {
        const int kb = q * 32;
        v8bf lo = *(const v8bf*)(arow + kb + half * 16);
        v8bf hi = *(const v8bf*)(arow + kb + half * 16 + 8);
#pragma unroll
        for (int i = 0; i < 8; ++i) { afr[q][i] = lo[i]; afr[q][8 + i] = hi[i]; }
        bfr[q] = *(const v16bf*)(brow + kb + half * 16);
    }

    v8f acc = {};
#pragma unroll
    for (int q = 0; q < 4; ++q)
        acc = wmma_bf16(afr[q], bfr[q], acc);

#pragma unroll
    for (int j = 0; j < 8; ++j) {
        const int rr = tm * 16 + half * 8 + j;
        const int cc = tn * 16 + r;
        if (rr < NPG && cc < NPG) {
            const long idx = ((long)g * NPG + rr) * NPG + cc;
            S[idx] = mask[idx] ? acc[j] : NEGV;
        }
    }
}

// ---------------------------------------------------------------------------
// Elementwise / reduction helpers
// ---------------------------------------------------------------------------
__global__ void wtrans(const float* __restrict__ W, __bf16* __restrict__ Wt, int K)
{   // W[K][128] -> Wt[128][K] bf16
    int i = blockIdx.x * blockDim.x + threadIdx.x;
    int n = HD * K;
    if (i >= n) return;
    int nc = i / K, k = i % K;
    Wt[i] = (__bf16)W[(long)k * HD + nc];
}

__global__ void deg_init(float* deg)
{
    int i = blockIdx.x * blockDim.x + threadIdx.x;
    if (i < NN) deg[i] = 1.0f;   // self-loop
}

__global__ void deg_acc(const int* __restrict__ ei, int E, float* __restrict__ deg)
{
    int e = blockIdx.x * blockDim.x + threadIdx.x;
    if (e >= E) return;
    atomicAdd(&deg[ei[E + e]], 1.0f);
}

__global__ void agg3_edges(const int* __restrict__ ei, int E,
                           const float* __restrict__ x, float* __restrict__ agg)
{
    int e = blockIdx.x * blockDim.x + threadIdx.x;
    if (e >= E) return;
    int s = ei[e], d = ei[E + e];
    atomicAdd(&agg[(long)d * 3 + 0], x[(long)s * 3 + 0]);
    atomicAdd(&agg[(long)d * 3 + 1], x[(long)s * 3 + 1]);
    atomicAdd(&agg[(long)d * 3 + 2], x[(long)s * 3 + 2]);
}

__global__ void aggH_edges(const int* __restrict__ ei, int E,
                           const float* __restrict__ h, float* __restrict__ agg)
{
    long tid = (long)blockIdx.x * blockDim.x + threadIdx.x;
    if (tid >= (long)E * 32) return;
    int e = (int)(tid >> 5), q = (int)(tid & 31);
    int s = ei[e], d = ei[E + e];
    float4 v = *(const float4*)(h + (long)s * HD + q * 4);
    float* a = agg + (long)d * HD + q * 4;
    atomicAdd(a + 0, v.x); atomicAdd(a + 1, v.y);
    atomicAdd(a + 2, v.z); atomicAdd(a + 3, v.w);
}

__global__ void make_zin3(const float* __restrict__ x, const float* __restrict__ agg,
                          const float* __restrict__ deg, float* __restrict__ zin, int n)
{
    int i = blockIdx.x * blockDim.x + threadIdx.x;
    if (i >= n) return;
    int row = i / 3;
    float inv = 1.0f / deg[row];
    zin[i] = x[i] * (1.0f + inv) + agg[i] * inv;
}

__global__ void make_zin(const float* __restrict__ h, const float* __restrict__ agg,
                         const float* __restrict__ deg, float* __restrict__ zin, int n)
{
    int i = blockIdx.x * blockDim.x + threadIdx.x;
    if (i >= n) return;
    int row = i >> 7;
    float inv = 1.0f / deg[row];
    zin[i] = h[i] * (1.0f + inv) + agg[i] * inv;
}

__global__ void gemm_in3(const float* __restrict__ zin, const float* __restrict__ W,
                         const float* __restrict__ b, float* __restrict__ Z)
{   // K = 3 input layer (scalar path; too thin for WMMA)
    int i = blockIdx.x * blockDim.x + threadIdx.x;   // NN*128
    int row = i >> 7, c = i & 127;
    const float* zr = zin + (long)row * 3;
    Z[i] = b[c] + zr[0] * W[c] + zr[1] * W[HD + c] + zr[2] * W[2 * HD + c];
}

__global__ void bn_stats(const float* __restrict__ Z, float* __restrict__ sums,
                         float* __restrict__ sumsq)
{   // grid 256 x block 128 ; 250 rows per block
    int f = threadIdx.x;
    long r0 = (long)blockIdx.x * 250;
    float s = 0.f, s2 = 0.f;
    for (int i = 0; i < 250; ++i) {
        float v = Z[(r0 + i) * HD + f];
        s += v; s2 += v * v;
    }
    atomicAdd(&sums[f], s);
    atomicAdd(&sumsq[f], s2);
}

__global__ void bn_final(const float* __restrict__ sums, const float* __restrict__ sumsq,
                         const float* __restrict__ g, const float* __restrict__ be,
                         float* __restrict__ scl, float* __restrict__ shf)
{
    int f = threadIdx.x;
    float mu  = sums[f]  * (1.0f / (float)NN);
    float var = sumsq[f] * (1.0f / (float)NN) - mu * mu;
    float sc  = g[f] * rsqrtf(var + 1e-5f);
    scl[f] = sc;
    shf[f] = be[f] - mu * sc;
}

__global__ void bn_relu(const float* __restrict__ Z, const float* __restrict__ scl,
                        const float* __restrict__ shf, float* __restrict__ R, int n)
{
    int i = blockIdx.x * blockDim.x + threadIdx.x;
    if (i >= n) return;
    int f = i & 127;
    float v = Z[i] * scl[f] + shf[f];
    R[i] = v > 0.f ? v : 0.f;
}

__global__ void gp_acc(const float* __restrict__ h, float* __restrict__ gp)
{   // grid NB x block 128: graph mean pooling, accumulated
    int g = blockIdx.x, f = threadIdx.x;
    const float* hr = h + (long)g * NPG * HD + f;
    float s = 0.f;
    for (int n = 0; n < NPG; ++n) s += hr[(long)n * HD];
    gp[g * HD + f] += s * (1.0f / NPG);
}

__global__ void concat_a0(const float* __restrict__ npool, const float* __restrict__ gp,
                          float* __restrict__ a0)
{
    long i = (long)blockIdx.x * blockDim.x + threadIdx.x;   // NN*256
    int c = (int)(i & 255);
    long row = i >> 8;
    float v = (c < HD) ? npool[row * HD + c]
                       : gp[(row / NPG) * HD + (c - HD)];
    a0[i] = v;
}

__global__ void pad_bf16(const float* __restrict__ a, __bf16* __restrict__ apad)
{
    long i = (long)blockIdx.x * blockDim.x + threadIdx.x;   // NB*512*128
    long f = i & 127, rowp = i >> 7;
    long g = rowp >> 9, rr = rowp & 511;
    __bf16 v = (__bf16)0.0f;
    if (rr < NPG) v = (__bf16)a[((long)g * NPG + rr) * HD + f];
    apad[i] = v;
}

__device__ __forceinline__ float u01(unsigned x)
{
    x ^= x >> 16; x *= 0x7feb352dU;
    x ^= x >> 15; x *= 0x846ca68bU;
    x ^= x >> 16;
    return ((float)(x >> 8) + 0.5f) * (1.0f / 16777216.0f);
}

__global__ void sample_reduce(const float* __restrict__ S, float* __restrict__ out)
{   // one block per graph: online logsumexp + Gumbel-argmax over 250000
    int b = blockIdx.x;
    const float* s = S + (long)b * (NPG * NPG);
    const int n = NPG * NPG;

    float m = -INFINITY, acc = 0.0f;
    float bp = -INFINITY; int bi = 0; float bx = 0.0f;
    for (int i = threadIdx.x; i < n; i += blockDim.x) {
        float x = s[i];
        if (x > m) { acc = acc * __expf(m - x) + 1.0f; m = x; }
        else       { acc += __expf(x - m); }
        float u  = u01((unsigned)(b * 1000003 + i));
        float gp = x - __logf(-__logf(u));
        if (gp > bp) { bp = gp; bi = i; bx = x; }
    }

    __shared__ float sm[256], sa[256], sp[256], sx[256];
    __shared__ int   si[256];
    int t = threadIdx.x;
    sm[t] = m; sa[t] = acc; sp[t] = bp; sx[t] = bx; si[t] = bi;
    __syncthreads();
    for (int off = 128; off > 0; off >>= 1) {
        if (t < off) {
            float m2 = sm[t + off], a2 = sa[t + off];
            float M = fmaxf(sm[t], m2);
            float A = (M == -INFINITY) ? 0.0f
                    : sa[t] * __expf(sm[t] - M) + a2 * __expf(m2 - M);
            sm[t] = M; sa[t] = A;
            if (sp[t + off] > sp[t]) { sp[t] = sp[t + off]; sx[t] = sx[t + off]; si[t] = si[t + off]; }
        }
        __syncthreads();
    }
    if (t == 0) {
        float lse = sm[0] + __logf(sa[0]);
        out[b]      = (float)si[0];     // action id
        out[NB + b] = sx[0] - lse;      // log prob
    }
}

// ---------------------------------------------------------------------------
// Host orchestration
// ---------------------------------------------------------------------------
extern "C" void kernel_launch(void* const* d_in, const int* in_sizes, int n_in,
                              void* d_out, int out_size, void* d_ws, size_t ws_size,
                              hipStream_t stream)
{
    const float* x    = (const float*)d_in[0];
    const int*   ei   = (const int*)d_in[1];
    const unsigned char* mask = (const unsigned char*)d_in[3];   // bool8
    const float* gW1_0 = (const float*)d_in[4];
    const float* gb1_0 = (const float*)d_in[5];
    const float* gg_0  = (const float*)d_in[6];
    const float* gbe_0 = (const float*)d_in[7];
    const float* gW2_0 = (const float*)d_in[8];
    const float* gb2_0 = (const float*)d_in[9];
    const float* gW1   = (const float*)d_in[10];
    const float* gb1   = (const float*)d_in[11];
    const float* gg    = (const float*)d_in[12];
    const float* gbe   = (const float*)d_in[13];
    const float* gW2   = (const float*)d_in[14];
    const float* gb2   = (const float*)d_in[15];
    const float* pW1_0 = (const float*)d_in[16];
    const float* pb1_0 = (const float*)d_in[17];
    const float* pW2_0 = (const float*)d_in[18];
    const float* pb2_0 = (const float*)d_in[19];
    const float* pW1   = (const float*)d_in[20];
    const float* pb1   = (const float*)d_in[21];
    const float* pW2   = (const float*)d_in[22];
    const float* pb2   = (const float*)d_in[23];

    const int E = in_sizes[1] / 2;

    // ---- workspace layout (floats) ------------------------------------
    float* W   = (float*)d_ws;
    float* AGG = W;                               // 16,384,000 (agg / a0)
    float* ZIN = AGG + (size_t)NN * 256;          // 8,192,000  (zin / t)
    float* ZZ  = ZIN + (size_t)NN * HD;           // 8,192,000  (z / a)
    float* RR  = ZZ  + (size_t)NN * HD;           // 8,192,000  (relu buf / apad)
    float* H0  = RR  + (size_t)NN * HD;
    float* H1  = H0  + (size_t)NN * HD;
    float* NP  = H1  + (size_t)NN * HD;
    float* GP  = NP  + (size_t)NN * HD;           // 16384
    float* DEG = GP  + (size_t)NB * HD;           // 64000
    float* SUMS  = DEG + NN;                      // 128
    float* SUMSQ = SUMS + HD;                     // 128
    float* SCL   = SUMSQ + HD;                    // 128
    float* SHF   = SCL + HD;                      // 128
    __bf16* WT   = (__bf16*)(SHF + HD);           // 229,376 bf16
    float* SCORES = (float*)(WT + 229376);        // 32,000,000
    __bf16* APAD  = (__bf16*)RR;                  // aliased (16.8MB <= 32MB)

    // transposed bf16 weight slots
    __bf16* wt_gW2_0 = WT;
    __bf16* wt_gW1l  = WT + 1 * 16384;            // 3 slots
    __bf16* wt_gW2l  = WT + 4 * 16384;            // 3 slots
    __bf16* wt_pW1_0 = WT + 7 * 16384;            // K=256, 32768 elems
    __bf16* wt_pW2_0 = WT + 7 * 16384 + 32768;
    __bf16* wt_pW1l  = WT + 8 * 16384 + 32768;    // 2 slots
    __bf16* wt_pW2l  = WT + 10 * 16384 + 32768;   // 2 slots

    const int EB = 256;
    #define GRD(n) (((n) + EB - 1) / EB)

    // ---- weight prep (f32 -> bf16 transposed) -------------------------
    wtrans<<<GRD(HD * HD), EB, 0, stream>>>(gW2_0, wt_gW2_0, HD);
    for (int l = 0; l < 3; ++l) {
        wtrans<<<GRD(HD * HD), EB, 0, stream>>>(gW1 + (size_t)l * HD * HD, wt_gW1l + (size_t)l * 16384, HD);
        wtrans<<<GRD(HD * HD), EB, 0, stream>>>(gW2 + (size_t)l * HD * HD, wt_gW2l + (size_t)l * 16384, HD);
    }
    wtrans<<<GRD(HD * 256), EB, 0, stream>>>(pW1_0, wt_pW1_0, 256);
    wtrans<<<GRD(HD * HD), EB, 0, stream>>>(pW2_0, wt_pW2_0, HD);
    for (int l = 0; l < 2; ++l) {
        wtrans<<<GRD(HD * HD), EB, 0, stream>>>(pW1 + (size_t)l * HD * HD, wt_pW1l + (size_t)l * 16384, HD);
        wtrans<<<GRD(HD * HD), EB, 0, stream>>>(pW2 + (size_t)l * HD * HD, wt_pW2l + (size_t)l * 16384, HD);
    }

    // ---- degrees ------------------------------------------------------
    deg_init<<<GRD(NN), EB, 0, stream>>>(DEG);
    deg_acc<<<GRD(E), EB, 0, stream>>>(ei, E, DEG);

    // ---- pooling accumulators -----------------------------------------
    hipMemsetAsync(NP, 0, (size_t)NN * HD * 4, stream);
    hipMemsetAsync(GP, 0, (size_t)NB * HD * 4, stream);

    // ---- GIN layer 0 (K=3 input) --------------------------------------
    hipMemsetAsync(AGG, 0, (size_t)NN * 3 * 4, stream);
    agg3_edges<<<GRD(E), EB, 0, stream>>>(ei, E, x, AGG);
    make_zin3<<<GRD(NN * 3), EB, 0, stream>>>(x, AGG, DEG, ZIN, NN * 3);
    gemm_in3<<<GRD(NN * HD), EB, 0, stream>>>(ZIN, gW1_0, gb1_0, ZZ);
    hipMemsetAsync(SUMS, 0, 2 * HD * 4, stream);
    bn_stats<<<256, 128, 0, stream>>>(ZZ, SUMS, SUMSQ);
    bn_final<<<1, 128, 0, stream>>>(SUMS, SUMSQ, gg_0, gbe_0, SCL, SHF);
    bn_relu<<<GRD(NN * HD), EB, 0, stream>>>(ZZ, SCL, SHF, RR, NN * HD);
    gemm_wmma_128<128, 0, 1><<<NN / 64, 128, 0, stream>>>(RR, wt_gW2_0, gb2_0, H0, NP);
    gp_acc<<<NB, 128, 0, stream>>>(H0, GP);

    // ---- GIN layers 1..3 ----------------------------------------------
    float* Hc = H0; float* Hn = H1;
    for (int l = 0; l < 3; ++l) {
        hipMemsetAsync(AGG, 0, (size_t)NN * HD * 4, stream);
        aggH_edges<<<GRD(E * 32), EB, 0, stream>>>(ei, E, Hc, AGG);
        make_zin<<<GRD(NN * HD), EB, 0, stream>>>(Hc, AGG, DEG, ZIN, NN * HD);
        gemm_wmma_128<128, 0, 0><<<NN / 64, 128, 0, stream>>>(ZIN, wt_gW1l + (size_t)l * 16384,
                                                              gb1 + (size_t)l * HD, ZZ, nullptr);
        hipMemsetAsync(SUMS, 0, 2 * HD * 4, stream);
        bn_stats<<<256, 128, 0, stream>>>(ZZ, SUMS, SUMSQ);
        bn_final<<<1, 128, 0, stream>>>(SUMS, SUMSQ, gg + (size_t)l * HD, gbe + (size_t)l * HD, SCL, SHF);
        bn_relu<<<GRD(NN * HD), EB, 0, stream>>>(ZZ, SCL, SHF, RR, NN * HD);
        gemm_wmma_128<128, 0, 1><<<NN / 64, 128, 0, stream>>>(RR, wt_gW2l + (size_t)l * 16384,
                                                              gb2 + (size_t)l * HD, Hn, NP);
        gp_acc<<<NB, 128, 0, stream>>>(Hn, GP);
        float* tmp = Hc; Hc = Hn; Hn = tmp;
    }

    // ---- policy MLP ----------------------------------------------------
    concat_a0<<<GRD(NN * 256), EB, 0, stream>>>(NP, GP, AGG);   // a0 in AGG
    gemm_wmma_128<256, 1, 0><<<NN / 64, 128, 0, stream>>>(AGG, wt_pW1_0, pb1_0, ZIN, nullptr);
    gemm_wmma_128<128, 0, 0><<<NN / 64, 128, 0, stream>>>(ZIN, wt_pW2_0, pb2_0, ZZ, nullptr);
    for (int l = 0; l < 2; ++l) {
        gemm_wmma_128<128, 1, 0><<<NN / 64, 128, 0, stream>>>(ZZ, wt_pW1l + (size_t)l * 16384,
                                                              pb1 + (size_t)l * HD, ZIN, nullptr);
        gemm_wmma_128<128, 0, 0><<<NN / 64, 128, 0, stream>>>(ZIN, wt_pW2l + (size_t)l * 16384,
                                                              pb2 + (size_t)l * HD, ZZ, nullptr);
    }

    // ---- scores + masked log-softmax + sample ---------------------------
    pad_bf16<<<GRD(NB * 512 * HD), EB, 0, stream>>>(ZZ, APAD);
    scores_wmma<<<(NB * 32 * 32) / 4, 128, 0, stream>>>(APAD, mask, SCORES);
    sample_reduce<<<NB, 256, 0, stream>>>(SCORES, (float*)d_out);

    (void)n_in; (void)out_size; (void)ws_size;
}
